// TemporalSelfAttentionEdgeIndexCreatorLayer_23837068492980
// MI455X (gfx1250) — matrix operand order repaired and checked by reading, not compiled, not run on hardware
//
#include <hip/hip_runtime.h>

// Problem constants (match reference)
#define BATCH 16
#define NPTS  1024
#define EMB   256
#define HEADS 8
#define HDIM  32
#define EDGES 64

typedef __attribute__((ext_vector_type(16))) _Float16 v16h;
typedef __attribute__((ext_vector_type(8)))  _Float16 v8h;
typedef __attribute__((ext_vector_type(8)))  float    v8f;

__device__ __forceinline__ v16h combine16(v8h lo, v8h hi) {
    v16h a;
#pragma unroll
    for (int i = 0; i < 8; i++) { a[i] = lo[i]; a[i + 8] = hi[i]; }
    return a;
}

// ---------------------------------------------------------------------------
// Kernel 1: per-head projection  Kp = K @ Wk^T,  Qp = Q @ Wq^T  (f32 -> f16)
// grid = (NPTS/16, BATCH), block = 256 (8 waves, one head per wave)
// WMMA: A = 16x32 f16 input rows (head slice), B = 32x32 W^T as two 16-col tiles
// ---------------------------------------------------------------------------
__global__ void proj_kernel(const float* __restrict__ keys,
                            const float* __restrict__ query,
                            const float* __restrict__ Wk,
                            const float* __restrict__ Wq,
                            _Float16* __restrict__ Kp,
                            _Float16* __restrict__ Qp)
{
    const int lane = threadIdx.x & 31;
    const int h    = threadIdx.x >> 5;        // head index, 8 waves
    const int row0 = blockIdx.x * 16;
    const int b    = blockIdx.y;

    // ---- A fragments (16x32, f16). contiguous k order used here:
    // lanes 0-15:  halves 0-15 -> K=0..15
    // lanes 16-31: halves 0-15 -> K=16..31
    const int m     = lane & 15;
    const int kbase = (lane < 16) ? 0 : 16;
    const size_t inoff = ((size_t)b * NPTS + row0 + m) * EMB + (size_t)h * HDIM;

    v16h aK, aQ;
#pragma unroll
    for (int i = 0; i < 8; i++) {
        aK[i]     = (_Float16)keys [inoff + kbase + i];
        aK[i + 8] = (_Float16)keys [inoff + kbase + 8 + i];
        aQ[i]     = (_Float16)query[inoff + kbase + i];
        aQ[i + 8] = (_Float16)query[inoff + kbase + 8 + i];
    }

    // ---- B fragments (32x16): B[d][e] = W[e][d]  (x @ W^T)
    // N = lane&15, K = half index + (lane<16 ? 0 : 16)  -> contiguous in W row e
    const int ecol  = lane & 15;
    const int dbase = (lane < 16) ? 0 : 16;
    const int mrow  = (lane < 16) ? 0 : 8;    // C/D layout: M = vgpr + mrow, N = lane&15

#pragma unroll
    for (int et = 0; et < 2; et++) {
        v16h bK, bQ;
        const float* wkr = Wk + (size_t)(et * 16 + ecol) * HDIM + dbase;
        const float* wqr = Wq + (size_t)(et * 16 + ecol) * HDIM + dbase;
#pragma unroll
        for (int i = 0; i < 16; i++) {
            bK[i] = (_Float16)wkr[i];
            bQ[i] = (_Float16)wqr[i];
        }
        v8f z = {0.f, 0.f, 0.f, 0.f, 0.f, 0.f, 0.f, 0.f};
        v8f dK = __builtin_amdgcn_wmma_f32_16x16x32_f16(false, aK, false, bK,
                                                        (short)0, z, false, false);
        v8f dQ = __builtin_amdgcn_wmma_f32_16x16x32_f16(false, aQ, false, bQ,
                                                        (short)0, z, false, false);
#pragma unroll
        for (int v = 0; v < 8; v++) {
            const size_t o = ((size_t)b * NPTS + row0 + mrow + v) * EMB
                           + (size_t)h * HDIM + et * 16 + ecol;
            Kp[o] = (_Float16)dK[v];
            Qp[o] = (_Float16)dQ[v];
        }
    }
}

// ---------------------------------------------------------------------------
// Kernel 2: S = Qp @ Kp^T per batch (16-row q tile x 1024 keys), then
// per-row top-64 (argmax selection, tie -> lowest index), sort indices
// ascending by rank counting, write edge_index.
// grid = (NPTS/16, BATCH), block = 512 (16 waves), dyn LDS = 16*1024*4 = 64 KB
// ---------------------------------------------------------------------------
__global__ void attn_topk_kernel(const _Float16* __restrict__ Qp,
                                 const _Float16* __restrict__ Kp,
                                 float* __restrict__ out)
{
    extern __shared__ float S[];              // [16][1024] score tile
    const int lane = threadIdx.x & 31;
    const int wave = threadIdx.x >> 5;        // 16 waves
    const int q0   = blockIdx.x * 16;
    const int b    = blockIdx.y;

    // A fragment addressing (Qp rows, f16)
    const int m     = lane & 15;
    const int kbase = (lane < 16) ? 0 : 16;
    const _Float16* qrow = Qp + ((size_t)b * NPTS + q0 + m) * EMB;

    // B fragment addressing: B[e][k] = Kp[k][e] -> row = key0 + (lane&15),
    // 16 contiguous halves per lane starting at dbase
    const int ncol  = lane & 15;
    const int dbase = (lane < 16) ? 0 : 16;
    const int mrow  = (lane < 16) ? 0 : 8;

    // ---- GEMM phase: each wave computes 4 key tiles of 16 columns
#pragma unroll
    for (int t = 0; t < 4; t++) {
        const int key0 = (wave * 4 + t) * 16;
        const _Float16* krow = Kp + ((size_t)b * NPTS + key0 + ncol) * EMB + dbase;
        __builtin_prefetch(krow, 0, 0);       // global_prefetch_b8
        v8f acc = {0.f, 0.f, 0.f, 0.f, 0.f, 0.f, 0.f, 0.f};
#pragma unroll
        for (int kk = 0; kk < 8; kk++) {      // K = 256 in 8 steps of 32
            v8h alo = *(const v8h*)(qrow + kk * 32 + kbase);
            v8h ahi = *(const v8h*)(qrow + kk * 32 + kbase + 8);
            v8h blo = *(const v8h*)(krow + kk * 32);
            v8h bhi = *(const v8h*)(krow + kk * 32 + 8);
            v16h a  = combine16(alo, ahi);
            v16h bb = combine16(blo, bhi);
            acc = __builtin_amdgcn_wmma_f32_16x16x32_f16(false, a, false, bb,
                                                         (short)0, acc, false, false);
        }
#pragma unroll
        for (int v = 0; v < 8; v++)
            S[(mrow + v) * NPTS + key0 + ncol] = acc[v];
    }
    __syncthreads();

    // ---- Top-64 of row `wave` (one wave per row). Each lane owns the
    // strided slots {j*32+lane : j=0..31}, cached in registers.
    const float* Srow = S + wave * NPTS;
    float vals[32];
#pragma unroll
    for (int j = 0; j < 32; j++) vals[j] = Srow[j * 32 + lane];

    unsigned used = 0u;                       // per-lane used-slot bitmask
    int win0 = 0x7FFFFFFF, win1 = 0x7FFFFFFF; // winners t=lane, t=lane+32

    for (int t = 0; t < EDGES; t++) {
        float bv = -3.402823e38f;
        int   bj = -1;
#pragma unroll
        for (int j = 0; j < 32; j++) {        // lane-local argmax, lowest j on tie
            if (!((used >> j) & 1u)) {
                float v = vals[j];
                if (v > bv) { bv = v; bj = j; }
            }
        }
        int bi = (bj < 0) ? 0x7FFFFFFF : (bj * 32 + lane);
#pragma unroll
        for (int off = 16; off > 0; off >>= 1) {  // wave32 argmax reduce
            float ov = __shfl_xor(bv, off, 32);
            int   oi = __shfl_xor(bi, off, 32);
            if (ov > bv || (ov == bv && oi < bi)) { bv = ov; bi = oi; }
        }
        if (lane == (bi & 31)) used |= (1u << (bi >> 5));   // retire winner slot
        if ((t & 31) == lane) { if (t < 32) win0 = bi; else win1 = bi; }
    }

    // ---- Ascending sort of 64 distinct indices by rank counting (shfl bcast)
    int r0 = 0, r1 = 0;
#pragma unroll
    for (int o = 0; o < 32; o++) {
        int w0 = __shfl(win0, o, 32);
        int w1 = __shfl(win1, o, 32);
        r0 += (w0 < win0) + (w1 < win0);
        r1 += (w0 < win1) + (w1 < win1);
    }

    // ---- Emit edge_index [B, 2, Q*E]
    const int    q    = q0 + wave;
    const size_t QE   = (size_t)NPTS * EDGES;
    const size_t base = (size_t)b * 2 * QE;
    out[base + (size_t)q * EDGES + lane]       = (float)q;   // channel 0: node id
    out[base + (size_t)q * EDGES + lane + 32]  = (float)q;
    out[base + QE + (size_t)q * EDGES + r0]    = (float)win0; // channel 1: sorted idx
    out[base + QE + (size_t)q * EDGES + r1]    = (float)win1;
}

// ---------------------------------------------------------------------------
extern "C" void kernel_launch(void* const* d_in, const int* in_sizes, int n_in,
                              void* d_out, int out_size, void* d_ws, size_t ws_size,
                              hipStream_t stream)
{
    const float* keys  = (const float*)d_in[0];
    const float* query = (const float*)d_in[1];
    const float* Wk    = (const float*)d_in[2];
    const float* Wq    = (const float*)d_in[3];
    float* out = (float*)d_out;

    // Workspace: f16 Qp then f16 Kp, 8 MB each (16 MB total)
    _Float16* Qp = (_Float16*)d_ws;
    _Float16* Kp = Qp + (size_t)BATCH * NPTS * EMB;

    dim3 grid(NPTS / 16, BATCH);
    proj_kernel<<<grid, 256, 0, stream>>>(keys, query, Wk, Wq, Kp, Qp);

    const size_t smem = (size_t)16 * NPTS * sizeof(float);   // 64 KB score tile
    attn_topk_kernel<<<grid, 512, smem, stream>>>(Qp, Kp, out);

    (void)in_sizes; (void)n_in; (void)out_size; (void)ws_size;
}
